// HiPPOLegSKernel_89893665505533
// MI455X (gfx1250) — matrix operand order repaired and checked by reading, not compiled, not run on hardware
//
#include <hip/hip_runtime.h>
#include <hip/hip_bf16.h>

// HiPPO-LegS SSM kernel for MI455X (gfx1250, wave32, WMMA).
//
// Math: K[d, 64*i + j] = (C_d . G_i) . (Abar^j . bbar_d), G_i = Abar^(64 i)
// This removes the L=4096 sequential scan: only ~126 sequential 64x64 matmuls
// (done with f32 WMMA for precision), everything else is parallel WMMA GEMMs.
//
// D_MODEL=1024, N=64, L=4096.  Output: float K[1024][1][4096].

typedef __attribute__((ext_vector_type(2)))  float    v2f;
typedef __attribute__((ext_vector_type(8)))  float    v8f;
typedef __attribute__((ext_vector_type(16))) _Float16 v16h;

#define NST   64
#define DMODEL 1024
#define LLEN  4096

// Workspace layout (floats)
#define WS_ABAR 0                      // 64*64
#define WS_BMAT 4096                   // 64*64  (dt * Minv)
#define WS_APOW 8192                   // 64 * 64*64  (Abar^j, j=0..63)
#define WS_GPOW (8192 + 64*4096)       // 64 * 64*64  (Abar^(64 i), i=0..63)
#define WS_BBAR (WS_GPOW + 64*4096)    // 1024*64

// ---------------------------------------------------------------------------
// Kernel 1: dt = exp(log_dt); M = I - dt/2 A; Gauss-Jordan inversion of M
// (64x64 in LDS); Abar = Minv * (I + dt/2 A); Bmat = dt * Minv.
// One block, 256 threads (plain VALU -- tiny problem).
// ---------------------------------------------------------------------------
__global__ void hippo_setup(const float* __restrict__ A,
                            const float* __restrict__ logdt,
                            float* __restrict__ abar,
                            float* __restrict__ bmat) {
  __shared__ float MI[64][128];   // [M | I] augmented
  __shared__ float P [64][64];
  __shared__ float fac[64];
  const int t = threadIdx.x;
  const float dt = __expf(logdt[0]);
  const float h  = 0.5f * dt;

  for (int idx = t; idx < 64 * 64; idx += 256) {
    const int r = idx >> 6, c = idx & 63;
    const float a   = A[idx];
    const float eye = (r == c) ? 1.0f : 0.0f;
    MI[r][c]      = eye - h * a;
    MI[r][64 + c] = eye;
    P[r][c]       = eye + h * a;
  }
  __syncthreads();

  for (int k = 0; k < 64; ++k) {
    const float inv = 1.0f / MI[k][k];
    __syncthreads();
    if (t < 128) MI[k][t] *= inv;      // normalize pivot row
    __syncthreads();
    if (t < 64) fac[t] = (t == k) ? 0.0f : MI[t][k];
    __syncthreads();
    for (int idx = t; idx < 64 * 128; idx += 256) {
      const int r = idx >> 7, c = idx & 127;
      if (r != k) MI[r][c] -= fac[r] * MI[k][c];
    }
    __syncthreads();
  }

  for (int idx = t; idx < 64 * 64; idx += 256) {
    const int r = idx >> 6, c = idx & 63;
    float s = 0.0f;
    for (int k = 0; k < 64; ++k) s += MI[r][64 + k] * P[k][c];
    abar[idx] = s;                      // Abar = Minv @ P
    bmat[idx] = dt * MI[r][64 + c];     // Bmat = dt * Minv
  }
}

// ---------------------------------------------------------------------------
// f32 WMMA 16x16 tile of a 64x64x64 matmul  D_tile = Asrc @ Bsrc
// (fragment layouts per CDNA5 ISA 7.12.2: 32-bit A 16x4,
//  B row-striped across lanes, C/D row = vgpr + 8*(lane>>4)).
// ---------------------------------------------------------------------------
__device__ __forceinline__ void mm64_tile_f32(const float* Asrc,
                                              const float* Bsrc,
                                              float* Dst,
                                              int mt, int nt, int lane) {
  const int lo = lane & 15, hi = lane >> 4;
  v8f c = {};
#pragma unroll
  for (int kk = 0; kk < 16; ++kk) {
    const int k0 = kk * 4 + 2 * hi;
    v2f a, b;
    a.x = Asrc[(mt * 16 + lo) * 64 + k0];
    a.y = Asrc[(mt * 16 + lo) * 64 + k0 + 1];
    b.x = Bsrc[(k0    ) * 64 + nt * 16 + lo];
    b.y = Bsrc[(k0 + 1) * 64 + nt * 16 + lo];
    c = __builtin_amdgcn_wmma_f32_16x16x4_f32(false, a, false, b,
                                              (short)0, c, false, false);
  }
#pragma unroll
  for (int v = 0; v < 8; ++v)
    Dst[(mt * 16 + v + 8 * hi) * 64 + nt * 16 + lo] = c[v];
}

// ---------------------------------------------------------------------------
// Kernel 2: power chains with f32 WMMA.  One block, 512 threads (16 waves:
// one 16x16 tile each).  apow[j] = Abar^j (j=0..63), gpow[i] = Abar^(64 i).
// ---------------------------------------------------------------------------
__global__ void hippo_powers(const float* __restrict__ abar_g,
                             float* __restrict__ apow,
                             float* __restrict__ gpow) {
  __shared__ float bufA[4096], bufB[4096], mul[4096];
  const int t = threadIdx.x, lane = t & 31, w = t >> 5;
  const int mt = w >> 2, nt = w & 3;

  for (int i = t; i < 4096; i += 512) {
    mul[i] = abar_g[i];
    const float eye = ((i >> 6) == (i & 63)) ? 1.0f : 0.0f;
    bufA[i] = eye;
    apow[i] = eye;                               // Abar^0 = I
  }
  __syncthreads();

  float* cur = bufA;
  float* nxt = bufB;
  for (int j = 1; j < 64; ++j) {                 // cur = Abar^(j-1) -> Abar^j
    mm64_tile_f32(cur, mul, nxt, mt, nt, lane);
    __syncthreads();
    float* tmp = cur; cur = nxt; nxt = tmp;
    for (int i = t; i < 4096; i += 512) apow[j * 4096 + i] = cur[i];
  }

  // A64 = Abar^63 @ Abar
  mm64_tile_f32(cur, mul, nxt, mt, nt, lane);
  __syncthreads();
  for (int i = t; i < 4096; i += 512) {
    mul[i] = nxt[i];                             // multiplier = Abar^64
    const float eye = ((i >> 6) == (i & 63)) ? 1.0f : 0.0f;
    cur[i]  = eye;
    gpow[i] = eye;                               // G_0 = I
  }
  __syncthreads();

  for (int i2 = 1; i2 < 64; ++i2) {              // G_i = G_{i-1} @ Abar^64
    mm64_tile_f32(cur, mul, nxt, mt, nt, lane);
    __syncthreads();
    float* tmp = cur; cur = nxt; nxt = tmp;
    for (int i = t; i < 4096; i += 512) gpow[i2 * 4096 + i] = cur[i];
  }
}

// ---------------------------------------------------------------------------
// Kernel 3: Bbar = B @ Bmat^T  (1024x64 = 1024x64 @ 64x64), f32 WMMA.
// Grid 64 blocks (16 rows each) x 128 threads (4 waves = 4 N-tiles).
// ---------------------------------------------------------------------------
__global__ void hippo_bbar(const float* __restrict__ B,
                           const float* __restrict__ bmat,
                           float* __restrict__ bbar) {
  const int t = threadIdx.x, lane = t & 31, lo = lane & 15, hi = lane >> 4;
  const int nt = t >> 5;
  const int dd0 = blockIdx.x * 16;
  v8f c = {};
#pragma unroll
  for (int kk = 0; kk < 16; ++kk) {
    const int k0 = kk * 4 + 2 * hi;
    v2f a, b;
    a.x = B[(dd0 + lo) * 64 + k0];
    a.y = B[(dd0 + lo) * 64 + k0 + 1];
    // B-operand element (k,n) = Bmat^T[k,n] = Bmat[n,k]
    b.x = bmat[(nt * 16 + lo) * 64 + k0];
    b.y = bmat[(nt * 16 + lo) * 64 + k0 + 1];
    c = __builtin_amdgcn_wmma_f32_16x16x4_f32(false, a, false, b,
                                              (short)0, c, false, false);
  }
#pragma unroll
  for (int v = 0; v < 8; ++v)
    bbar[(dd0 + v + 8 * hi) * 64 + nt * 16 + lo] = c[v];
}

// ---------------------------------------------------------------------------
// Kernel 4: main.  grid = (64 d-tiles, 4 i-chunks), 256 threads (8 waves).
//  Phase A: CT[il][dd][n] = (C_tile @ G_i^T)  for 16 i's  -> LDS (f16, 32KB)
//  Phase B: RB[jl][dd][n] = (Bbar_tile @ Abar^j^T) for 16 j's -> LDS (f16)
//  Phase C: per channel dd: K_tile(16i x 16j) = CT_dd @ RB_dd^T via
//           v_wmma_f32_16x16x32_f16 (K=64 -> 2 steps), write K, add D at l=0.
// ---------------------------------------------------------------------------
__global__ void hippo_main(const float* __restrict__ C,
                           const float* __restrict__ Dvec,
                           const float* __restrict__ bbar,
                           const float* __restrict__ apow,
                           const float* __restrict__ gpow,
                           float* __restrict__ out) {
  __shared__ _Float16 CTs[16][16][64];  // [i_local][dd][n]  32 KB
  __shared__ _Float16 RBs[16][16][64];  // [j_local][dd][n]  32 KB
  const int t = threadIdx.x, lane = t & 31, lo = lane & 15, hi = lane >> 4;
  const int w = t >> 5;
  const int dd0 = blockIdx.x * 16;
  const int ic  = blockIdx.y;

  // ---- Phase A: CT chunk (16 i's x 4 n-tiles = 64 tile jobs over 8 waves)
  for (int job = w; job < 64; job += 8) {
    const int il = job >> 2, nt = job & 3;
    const float* G = gpow + (ic * 16 + il) * 4096;
    v8f c = {};
#pragma unroll
    for (int kk = 0; kk < 16; ++kk) {
      const int k0 = kk * 4 + 2 * hi;
      v2f a, b;
      a.x = C[(dd0 + lo) * 64 + k0];
      a.y = C[(dd0 + lo) * 64 + k0 + 1];
      b.x = G[(nt * 16 + lo) * 64 + k0];      // G_i^T[k,n] = G_i[n,k]
      b.y = G[(nt * 16 + lo) * 64 + k0 + 1];
      c = __builtin_amdgcn_wmma_f32_16x16x4_f32(false, a, false, b,
                                                (short)0, c, false, false);
    }
#pragma unroll
    for (int v = 0; v < 8; ++v)
      CTs[il][v + 8 * hi][nt * 16 + lo] = (_Float16)c[v];
  }
  __syncthreads();

  for (int jc = 0; jc < 4; ++jc) {
    if (jc) __syncthreads();                 // protect RBs vs prev Phase C

    // ---- Phase B: RB chunk
    for (int job = w; job < 64; job += 8) {
      const int jl = job >> 2, nt = job & 3;
      const float* Ap = apow + (jc * 16 + jl) * 4096;
      v8f c = {};
#pragma unroll
      for (int kk = 0; kk < 16; ++kk) {
        const int k0 = kk * 4 + 2 * hi;
        v2f a, b;
        a.x = bbar[(dd0 + lo) * 64 + k0];
        a.y = bbar[(dd0 + lo) * 64 + k0 + 1];
        b.x = Ap[(nt * 16 + lo) * 64 + k0];   // (Abar^j)^T[k,n] = Abar^j[n,k]
        b.y = Ap[(nt * 16 + lo) * 64 + k0 + 1];
        c = __builtin_amdgcn_wmma_f32_16x16x4_f32(false, a, false, b,
                                                  (short)0, c, false, false);
      }
#pragma unroll
      for (int v = 0; v < 8; ++v)
        RBs[jl][v + 8 * hi][nt * 16 + lo] = (_Float16)c[v];
    }
    __syncthreads();

    // ---- Phase C: per-channel contraction with f16 WMMA (K=64 -> 2 steps)
    for (int dd = w; dd < 16; dd += 8) {
      v8f acc = {};
#pragma unroll
      for (int s = 0; s < 2; ++s) {
        v16h a, b;
        // A fragment: 16x32 f16 (M = i_local, K = n).  VGPR v<4: K=2v(+8*hi),
        // v>=4: K=16+2(v-4)(+8*hi); pairs packed per VGPR.
        const int nb = 32 * s + 8 * hi;
#pragma unroll
        for (int v = 0; v < 4; ++v) {
          a[2 * v]     = CTs[lo][dd][nb + 2 * v];
          a[2 * v + 1] = CTs[lo][dd][nb + 2 * v + 1];
          a[2 * v + 8] = CTs[lo][dd][nb + 16 + 2 * v];
          a[2 * v + 9] = CTs[lo][dd][nb + 16 + 2 * v + 1];
        }
        // B fragment: 32x16 f16 (K = n, N = j).  VGPR v<4: K=2v(+8*hi),
        // v>=4: K=16+2(v-4)(+8*hi).  B[n,j] = RB[j][dd][n].
        const int kb = 32 * s + 8 * hi;
#pragma unroll
        for (int v = 0; v < 4; ++v) {
          b[2 * v]     = RBs[lo][dd][kb + 2 * v];
          b[2 * v + 1] = RBs[lo][dd][kb + 2 * v + 1];
          b[2 * v + 8] = RBs[lo][dd][kb + 16 + 2 * v];
          b[2 * v + 9] = RBs[lo][dd][kb + 16 + 2 * v + 1];
        }
        acc = __builtin_amdgcn_wmma_f32_16x16x32_f16(false, a, false, b,
                                                     (short)0, acc,
                                                     false, false);
      }
      const int d = dd0 + dd;
#pragma unroll
      for (int v = 0; v < 8; ++v) {
        const int ig = ic * 16 + v + 8 * hi;
        const int jg = jc * 16 + lo;
        float val = acc[v];
        if (ig == 0 && jg == 0) val += Dvec[d];   // K[:,0] += D
        out[d * LLEN + ig * 64 + jg] = val;
      }
    }
  }
}

// ---------------------------------------------------------------------------
extern "C" void kernel_launch(void* const* d_in, const int* in_sizes, int n_in,
                              void* d_out, int out_size, void* d_ws, size_t ws_size,
                              hipStream_t stream) {
  (void)in_sizes; (void)n_in; (void)out_size; (void)ws_size;
  const float* A_ct   = (const float*)d_in[0];   // 64*64
  const float* B      = (const float*)d_in[1];   // 1024*64
  const float* C      = (const float*)d_in[2];   // 1024*64
  const float* Dvec   = (const float*)d_in[3];   // 1024
  const float* log_dt = (const float*)d_in[4];   // 1

  float* ws   = (float*)d_ws;
  float* abar = ws + WS_ABAR;
  float* bmat = ws + WS_BMAT;
  float* apow = ws + WS_APOW;
  float* gpow = ws + WS_GPOW;
  float* bbar = ws + WS_BBAR;
  float* K    = (float*)d_out;

  hipLaunchKernelGGL(hippo_setup,  dim3(1),        dim3(256), 0, stream,
                     A_ct, log_dt, abar, bmat);
  hipLaunchKernelGGL(hippo_powers, dim3(1),        dim3(512), 0, stream,
                     abar, apow, gpow);
  hipLaunchKernelGGL(hippo_bbar,   dim3(64),       dim3(128), 0, stream,
                     B, bmat, bbar);
  hipLaunchKernelGGL(hippo_main,   dim3(64, 4),    dim3(256), 0, stream,
                     C, Dvec, bbar, apow, gpow, K);
}
